// RelativeNeuralNetworkAttentionMultiHeadScore_69020124447484
// MI455X (gfx1250) — matrix operand order repaired and checked by reading, not compiled, not run on hardware
//
#include <hip/hip_runtime.h>

// RelativeNeuralNetworkAttentionMultiHeadScore — CDNA5 / gfx1250
// B=8, T=128, Q=64, K=64, H=4, n=127. One block per (b,t): 1024 blocks x 256 threads.

typedef __attribute__((ext_vector_type(16))) _Float16 v16h;
typedef __attribute__((ext_vector_type(8)))  _Float16 v8h;
typedef __attribute__((ext_vector_type(4)))  _Float16 v4h;
typedef __attribute__((ext_vector_type(2)))  _Float16 v2h;
typedef __attribute__((ext_vector_type(8)))  float    v8f;

constexpr int NB  = 127;   // neighbors
constexpr int NR  = 128;   // padded rows
constexpr int LDA = 72;    // LDS row stride in halves (144 B: 16B-aligned, conflict-free)

// Branch-free tanh: hardware v_tanh_f32 if available, else via v_exp_f32.
__device__ inline float fast_tanh(float x) {
#if __has_builtin(__builtin_amdgcn_tanhf)
    return __builtin_amdgcn_tanhf(x);
#else
    float e = __expf(2.f * x);
    return 1.f - 2.f / (e + 1.f);
#endif
}

#define DPP8_SEL(a,b,c,d,e,f,g,h) \
    ((a)|((b)<<3)|((c)<<6)|((d)<<9)|((e)<<12)|((f)<<15)|((g)<<18)|((h)<<21))

#if __has_builtin(__builtin_amdgcn_mov_dpp8) && __has_builtin(__builtin_amdgcn_update_dpp)
#define USE_DPP_REDUCE 1

__device__ inline float dpp_xor1(float x) {
    return __int_as_float(__builtin_amdgcn_mov_dpp8(__float_as_int(x),
                          DPP8_SEL(1,0,3,2,5,4,7,6)));
}
__device__ inline float dpp_xor2(float x) {
    return __int_as_float(__builtin_amdgcn_mov_dpp8(__float_as_int(x),
                          DPP8_SEL(2,3,0,1,6,7,4,5)));
}
__device__ inline float dpp_xor4(float x) {
    return __int_as_float(__builtin_amdgcn_mov_dpp8(__float_as_int(x),
                          DPP8_SEL(4,5,6,7,0,1,2,3)));
}
__device__ inline float dpp_mirror16(float x) {   // lane L <-> 15-L within row of 16
    return __int_as_float(__builtin_amdgcn_update_dpp(0, __float_as_int(x),
                          0x140 /*row_mirror*/, 0xf, 0xf, true));
}

// Multi-value butterfly combine: merge two row-streams, summing over one column
// bit. This lane keeps stream (b ? hi : lo) and receives the same stream from
// its xor-partner (whose b is inverted, so the partner sends what we keep).
__device__ inline float combine1(float lo, float hi, bool b) {
    float send = b ? lo : hi, keep = b ? hi : lo;
    return keep + dpp_xor1(send);
}
__device__ inline float combine2(float lo, float hi, bool b) {
    float send = b ? lo : hi, keep = b ? hi : lo;
    return keep + dpp_xor2(send);
}
__device__ inline float combine4(float lo, float hi, bool b) {
    float send = b ? lo : hi, keep = b ? hi : lo;
    return keep + dpp_xor4(send);
}

// Full-wave (32 lane) reductions: DPP within 16, one shuffle across halves.
__device__ inline float wave32_sum(float x) {
    x += dpp_xor1(x); x += dpp_xor2(x); x += dpp_xor4(x); x += dpp_mirror16(x);
    x += __shfl_xor(x, 16, 32);
    return x;
}
__device__ inline float wave32_max(float x) {
    x = fmaxf(x, dpp_xor1(x)); x = fmaxf(x, dpp_xor2(x));
    x = fmaxf(x, dpp_xor4(x)); x = fmaxf(x, dpp_mirror16(x));
    x = fmaxf(x, __shfl_xor(x, 16, 32));
    return x;
}
#else
__device__ inline float wave32_sum(float x) {
    x += __shfl_xor(x, 1);  x += __shfl_xor(x, 2);  x += __shfl_xor(x, 4);
    x += __shfl_xor(x, 8);  x += __shfl_xor(x, 16);
    return x;
}
__device__ inline float wave32_max(float x) {
    x = fmaxf(x, __shfl_xor(x, 1));  x = fmaxf(x, __shfl_xor(x, 2));
    x = fmaxf(x, __shfl_xor(x, 4));  x = fmaxf(x, __shfl_xor(x, 8));
    x = fmaxf(x, __shfl_xor(x, 16));
    return x;
}
#endif

// ---- B fragment (32x16, f16) for v_wmma_f32_16x16x32_f16, loaded from Wk (row-major 256x64).
// lanes 0-15: h[i] = B[kk+i][n0+lane]; lanes 16-31: h[i] = B[kk+16+i][n0+lane-16].
// B[k][o] == Wk[o][k] -> each lane reads 16 contiguous floats of one Wk row.
__device__ inline v16h load_bfrag(const float* __restrict__ W, int n0, int kk, int lane) {
    int n  = n0 + (lane & 15);
    int kb = kk + ((lane & 16) ? 16 : 0);
    const float4* p = (const float4*)(W + n * 64 + kb);
    v16h r;
#pragma unroll
    for (int j = 0; j < 4; ++j) {
        float4 f = p[j];
        r[j * 4 + 0] = (_Float16)f.x;
        r[j * 4 + 1] = (_Float16)f.y;
        r[j * 4 + 2] = (_Float16)f.z;
        r[j * 4 + 3] = (_Float16)f.w;
    }
    return r;
}

// ---- A fragment (16x32, f16) from LDS. lanes 0-15 row M=lane,
// h[0..15]=K[kk+0..15]; lanes 16-31 same rows, K offset +16.
__device__ inline v16h load_afrag(const _Float16* __restrict__ Ain, int mt, int kk, int lane) {
    int r    = mt * 16 + (lane & 15);
    int base = r * LDA + kk + ((lane & 16) ? 16 : 0);
    v8h a0 = *(const v8h*)(Ain + base);
    v8h a1 = *(const v8h*)(Ain + base + 8);
    v16h out;
#pragma unroll
    for (int j = 0; j < 8; ++j) { out[j] = a0[j]; out[8 + j] = a1[j]; }
    return out;
}

__global__ __launch_bounds__(256)
void rnn_attn_mhs_kernel(const float* __restrict__ q_x,
                         const float* __restrict__ kv_x,
                         const float* __restrict__ Wk,
                         const float* __restrict__ Wq,
                         const float* __restrict__ Wv,
                         const float* __restrict__ bv,
                         const float* __restrict__ bias,
                         const float* __restrict__ ws,
                         const float* __restrict__ bs,
                         float* __restrict__ out)
{
    __shared__ __align__(16) _Float16 Ain[NR * LDA]; // kv tile, f16
    __shared__ float qb[256];                        // Wq@q + bias (per out channel)
    __shared__ float qrow[64];
    __shared__ float wsS[64];
    __shared__ float scoreS[NR * 4];                 // additive scores (n, h)
    __shared__ __align__(16) float wS[NR * 4];       // softmax weights
    __shared__ float AwS[4 * 64];                    // softmax-weighted kv reduction per head

    const int tid = threadIdx.x;
    const int bt  = blockIdx.x;                      // flat (b, t)
    const float* qp  = q_x  + (size_t)bt * 64;
    const float* kvp = kv_x + (size_t)bt * (size_t)(NB * 64);

    // ---- Stage 0: parallel staging ---------------------------------------
    if (tid < 64) { qrow[tid] = qp[tid]; wsS[tid] = ws[tid]; }
    AwS[tid] = 0.f;
    {
        const float b0 = *bs;
        for (int i = tid; i < NR * 4; i += 256) scoreS[i] = b0;
    }
    // kv (127x64 f32) -> LDS f16, row 127 zero-padded
    for (int i = tid; i < NR * 16; i += 256) {
        int r = i >> 4, c4 = i & 15;
        float4 f = make_float4(0.f, 0.f, 0.f, 0.f);
        if (r < NB) f = *(const float4*)(kvp + r * 64 + c4 * 4);
        v4h h;
        h[0] = (_Float16)f.x; h[1] = (_Float16)f.y;
        h[2] = (_Float16)f.z; h[3] = (_Float16)f.w;
        *(v4h*)(Ain + r * LDA + c4 * 4) = h;
    }
    __syncthreads();

    // ---- Stage 1: query projection qb[o] = (Wq @ q)[o] + bias[o&63] ------
    {
        const int o = tid;
        const float4* wr = (const float4*)(Wq + o * 64);
        float acc = 0.f;
#pragma unroll
        for (int j = 0; j < 16; ++j) {
            float4 f = wr[j];
            acc += f.x * qrow[j * 4 + 0] + f.y * qrow[j * 4 + 1]
                 + f.z * qrow[j * 4 + 2] + f.w * qrow[j * 4 + 3];
        }
        qb[o] = acc + bias[o & 63];
    }
    __syncthreads();

    // ---- Stage 2: keys = kv @ Wk^T via WMMA, fused tanh/score reduction --
    // wave w owns output cols [32w, 32w+32) -> single head (w>>1).
    {
        const int lane   = tid & 31;
        const int wv     = tid >> 5;
        const int n0a    = wv * 32;
        const int n0b    = n0a + 16;
        const int head   = wv >> 1;
        const int nn     = lane & 15;
        const int hiHalf = (lane & 16) ? 8 : 0;

        const v16h B00 = load_bfrag(Wk, n0a, 0,  lane);
        const v16h B01 = load_bfrag(Wk, n0a, 32, lane);
        const v16h B10 = load_bfrag(Wk, n0b, 0,  lane);
        const v16h B11 = load_bfrag(Wk, n0b, 32, lane);

        const float qbA = qb[n0a + nn], qbB = qb[n0b + nn];
        const float wsA = wsS[(n0a + nn) & 63], wsB = wsS[(n0b + nn) & 63];

#if USE_DPP_REDUCE
        // Keep-bits XOR'd with bit3 so row_mirror partners (nn <-> 15-nn) hold
        // the SAME row: row(e) = (nn&7) ^ (bit3 ? 7 : 0). Lanes nn<8 own rows 0-7.
        const bool f3  = (nn & 8) != 0;
        const bool s0b = ((nn & 1) != 0) != f3;
        const bool s1b = ((nn & 2) != 0) != f3;
        const bool s2b = ((nn & 4) != 0) != f3;
#endif

        for (int mt = 0; mt < 8; ++mt) {
            v16h A0 = load_afrag(Ain, mt, 0,  lane);
            v16h A1 = load_afrag(Ain, mt, 32, lane);
            v8f acc0 = {}, acc1 = {};
            acc0 = __builtin_amdgcn_wmma_f32_16x16x32_f16(false, A0, false, B00, (short)0, acc0, false, false);
            acc0 = __builtin_amdgcn_wmma_f32_16x16x32_f16(false, A1, false, B01, (short)0, acc0, false, false);
            acc1 = __builtin_amdgcn_wmma_f32_16x16x32_f16(false, A0, false, B10, (short)0, acc1, false, false);
            acc1 = __builtin_amdgcn_wmma_f32_16x16x32_f16(false, A1, false, B11, (short)0, acc1, false, false);

            // D layout: lanes 0-15 -> M=v, N=lane; lanes 16-31 -> M=8+v, N=lane-16.
            float cv[8];
#pragma unroll
            for (int v = 0; v < 8; ++v)
                cv[v] = fast_tanh(acc0[v] + qbA) * wsA
                      + fast_tanh(acc1[v] + qbB) * wsB;

            float tot;
#if USE_DPP_REDUCE
            // Multi-value butterfly: 3 DPP8 combine levels + 1 DPP16 mirror.
            float a0 = combine1(cv[0], cv[1], s0b);
            float a1 = combine1(cv[2], cv[3], s0b);
            float a2 = combine1(cv[4], cv[5], s0b);
            float a3 = combine1(cv[6], cv[7], s0b);
            float d0 = combine2(a0, a1, s1b);
            float d1 = combine2(a2, a3, s1b);
            float e  = combine4(d0, d1, s2b);
            tot = e + dpp_mirror16(e);          // lanes nn<8: tot = row nn sum
#else
            float myC = 0.f;
#pragma unroll
            for (int v = 0; v < 8; ++v) {
                float c = cv[v];
                c += __shfl_xor(c, 1, 16);
                c += __shfl_xor(c, 2, 16);
                c += __shfl_xor(c, 4, 16);
                c += __shfl_xor(c, 8, 16);
                myC = (nn == v) ? c : myC;
            }
            tot = myC;
#endif
            if (nn < 8) {                        // one predicated LDS atomic per row-tile
                int row = mt * 16 + nn + hiHalf;
                atomicAdd(&scoreS[row * 4 + head], tot);   // ds_add_f32
            }
        }
    }
    __syncthreads();

    // ---- Stage 3: softmax over neighbors — one wave per head -------------
    if (tid < 128) {
        const int h = tid >> 5, j = tid & 31;   // wave h handles head h
        float s0 = scoreS[(j      ) * 4 + h];
        float s1 = scoreS[(j + 32 ) * 4 + h];
        float s2 = scoreS[(j + 64 ) * 4 + h];
        float s3 = (j == 31) ? -3.0e38f : scoreS[(j + 96) * 4 + h]; // mask pad row 127
        float mx = wave32_max(fmaxf(fmaxf(s0, s1), fmaxf(s2, s3)));
        float e0 = __expf(s0 - mx), e1 = __expf(s1 - mx);
        float e2 = __expf(s2 - mx);
        float e3 = (j == 31) ? 0.f : __expf(s3 - mx);
        float inv = 1.f / wave32_sum(e0 + e1 + e2 + e3);
        wS[(j      ) * 4 + h] = e0 * inv;
        wS[(j + 32 ) * 4 + h] = e1 * inv;
        wS[(j + 64 ) * 4 + h] = e2 * inv;
        wS[(j + 96 ) * 4 + h] = e3 * inv;       // row 127 -> 0
    }
    __syncthreads();

    // ---- Stage 4: Aw[h][c] = sum_n w[n,h] * kv[n,c]  (values matmul folded) ----
    // Exact: x[o] = Aw[h(o)] . Wv[o,0:64] + q . Wv[o,64:128] + bv[o]
    // (softmax weights sum to 1, broadcast-q part of Vin identical over n).
    // Wave w reduces rows [16w, 16w+16); each lane owns a column pair; 8
    // register partials (4 heads x 2 cols) committed with ds_add_f32.
    {
        const int lane  = tid & 31;
        const int wv    = tid >> 5;
        const int c0    = lane * 2;
        const int nbase = wv * 16;
        float acc[4][2] = {};
#pragma unroll 4
        for (int i = 0; i < 16; ++i) {
            const int n = nbase + i;                       // row 127 has w = 0
            v2h    kv2 = *(const v2h*)(Ain + n * LDA + c0);
            float4 wn  = *(const float4*)(wS + n * 4);     // broadcast read
            float x0 = (float)kv2[0], x1 = (float)kv2[1];
            acc[0][0] += wn.x * x0;  acc[0][1] += wn.x * x1;
            acc[1][0] += wn.y * x0;  acc[1][1] += wn.y * x1;
            acc[2][0] += wn.z * x0;  acc[2][1] += wn.z * x1;
            acc[3][0] += wn.w * x0;  acc[3][1] += wn.w * x1;
        }
#pragma unroll
        for (int h = 0; h < 4; ++h) {
            atomicAdd(&AwS[h * 64 + c0    ], acc[h][0]);
            atomicAdd(&AwS[h * 64 + c0 + 1], acc[h][1]);
        }
    }
    __syncthreads();

    // ---- Stage 5: output projection --------------------------------------
    {
        const int o = tid, h = o >> 6;
        const float4* wr = (const float4*)(Wv + o * 128);
        float acc = bv[o];
#pragma unroll
        for (int j = 0; j < 16; ++j) {   // kv half of the concat
            float4 f = wr[j];
            acc += f.x * AwS[h * 64 + j * 4 + 0] + f.y * AwS[h * 64 + j * 4 + 1]
                 + f.z * AwS[h * 64 + j * 4 + 2] + f.w * AwS[h * 64 + j * 4 + 3];
        }
#pragma unroll
        for (int j = 0; j < 16; ++j) {   // broadcast-q half of the concat
            float4 f = wr[16 + j];
            acc += f.x * qrow[j * 4 + 0] + f.y * qrow[j * 4 + 1]
                 + f.z * qrow[j * 4 + 2] + f.w * qrow[j * 4 + 3];
        }
        out[(size_t)bt * 256 + o] = acc;
    }
}

extern "C" void kernel_launch(void* const* d_in, const int* in_sizes, int n_in,
                              void* d_out, int out_size, void* d_ws, size_t ws_size,
                              hipStream_t stream) {
    (void)in_sizes; (void)n_in; (void)out_size; (void)d_ws; (void)ws_size;
    const float* q_x  = (const float*)d_in[0];
    const float* kv_x = (const float*)d_in[1];
    const float* Wk   = (const float*)d_in[2];
    const float* Wq   = (const float*)d_in[3];
    const float* Wv   = (const float*)d_in[4];
    const float* bv   = (const float*)d_in[5];
    const float* bias = (const float*)d_in[6];
    const float* ws   = (const float*)d_in[7];
    const float* bs   = (const float*)d_in[8];

    dim3 grid(8 * 128);   // one block per (b, t)
    dim3 block(256);      // 8 wave32
    rnn_attn_mhs_kernel<<<grid, block, 0, stream>>>(
        q_x, kv_x, Wk, Wq, Wv, bv, bias, ws, bs, (float*)d_out);
}
